// BinSimGNN_41025527612068
// MI455X (gfx1250) — matrix operand order repaired and checked by reading, not compiled, not run on hardware
//
#include <hip/hip_runtime.h>

#define DIMC 128
#define HC 4
#define DC 32
#define FC 1000
#define BINSC 16
#define TC 16

typedef __attribute__((ext_vector_type(16))) __bf16 v16bf;
typedef __attribute__((ext_vector_type(8)))  float  v8f;

__device__ __forceinline__ unsigned short bsg_f2bf(float f) {
  union { float f; unsigned int u; } c; c.f = f;
  unsigned int u = c.u;
  unsigned int r = u + 0x7FFFu + ((u >> 16) & 1u);
  return (unsigned short)(r >> 16);
}
__device__ __forceinline__ unsigned int bsg_encf(float f) {
  union { float f; unsigned int u; } c; c.f = f;
  return (c.u & 0x80000000u) ? ~c.u : (c.u | 0x80000000u);
}
__device__ __forceinline__ float bsg_decf(unsigned int e) {
  union { float f; unsigned int u; } c;
  c.u = (e & 0x80000000u) ? (e & 0x7FFFFFFFu) : ~e;
  return c.f;
}
__device__ __forceinline__ float bsg_gelu(float x) {
  return 0.5f * x * (1.f + tanhf(0.7978845608028654f * (x + 0.044715f * x * x * x)));
}
__device__ __forceinline__ float bsg_sigm(float x) { return 1.f / (1.f + expf(-x)); }

// ---- wave32 WMMA 16x16 tile over K=128 from LDS bf16 tiles -----------------
// As: plain [16 rows][128 k] (A-fragment lanes read contiguous runs already).
//   A layout (16-bit 16x32): lane m=L&15, kbase=(L>>4)*16, e[j]=K kbase+j
//   -> one contiguous 32B run per k-step.
// Bs: FRAGMENT-ORDER pack [kk][col][kr]: Bs[kk*4096 + col*32 + kr], kr in [0,32).
//   B layout (16-bit 32x16): lane n=L&15, kbase=(L>>4)*16, e[j]=K kbase+j
//   -> one contiguous 32B run per lane per k-step (2x ds_load_b128).
__device__ __forceinline__ v8f bsg_wmma_tile(const unsigned short* As,
                                             const unsigned short* Bs,
                                             int lane, int col0) {
  v8f acc = {0.f, 0.f, 0.f, 0.f, 0.f, 0.f, 0.f, 0.f};
  const int mA = lane & 15, kbA = (lane >> 4) * 16;
  const int nB = col0 + (lane & 15), kbB = (lane >> 4) * 16;
  #pragma unroll
  for (int kk = 0; kk < 4; ++kk) {
    union { v16bf v; unsigned short e[16]; } af, bb;
    const unsigned short* ap = As + mA * 128 + kk * 32 + kbA;
    #pragma unroll
    for (int j = 0; j < 8; ++j) {
      af.e[j]     = ap[j];        // K = kb+kbA .. +7
      af.e[8 + j] = ap[8 + j];    // K = kb+kbA+8 .. +15
    }
    const unsigned short* bp = Bs + kk * 4096 + nB * 32 + kbB;
    #pragma unroll
    for (int j = 0; j < 16; ++j) bb.e[j] = bp[j];
    acc = __builtin_amdgcn_wmma_f32_16x16x32_bf16(false, af.v, false, bb.v,
                                                  (short)0, acc, false, false);
  }
  return acc;
}

// C[M x 128] = A[M x 128] @ W[128 x 128] (+ bias). mode1: A'=gelu(A),
// C = s*(A'@W+b) + (1-s)*X with s = sigmoid(*skipP). A/X rows split across two ptrs.
__global__ __launch_bounds__(256) void bsg_gemm128(
    const float* __restrict__ A1, int split, const float* __restrict__ A2,
    const float* __restrict__ W, const float* __restrict__ bias,
    float* __restrict__ C, int M, int mode, const float* __restrict__ skipP,
    const float* __restrict__ X1, const float* __restrict__ X2) {
  __shared__ unsigned short Bs[128 * 128];
  __shared__ unsigned short As[16 * 128];
  const int tid = threadIdx.x;
  const int m0 = blockIdx.x * 16;
  // W is [k][n] row-major; iterate with n (contiguous in global) innermost.
  for (int idx = tid; idx < 128 * 128; idx += 256) {
    int kk = idx >> 12, kr = (idx >> 7) & 31, col = idx & 127;
    Bs[kk * 4096 + col * 32 + kr] = bsg_f2bf(W[(size_t)(kk * 32 + kr) * 128 + col]);
  }
  for (int idx = tid; idx < 16 * 128; idx += 256) {
    int r = idx >> 7, c = idx & 127;
    int row = m0 + r;
    float v = 0.f;
    if (row < M) {
      const float* src = (row < split) ? (A1 + (size_t)row * 128)
                                       : (A2 + (size_t)(row - split) * 128);
      v = src[c];
      if (mode == 1) v = bsg_gelu(v);
    }
    As[idx] = bsg_f2bf(v);
  }
  __syncthreads();
  const int wave = tid >> 5, lane = tid & 31;
  const int col0 = wave * 16;
  v8f acc = bsg_wmma_tile(As, Bs, lane, col0);
  float s = 1.f, om = 0.f;
  if (mode == 1) { s = bsg_sigm(skipP[0]); om = 1.f - s; }
  const int col = col0 + (lane & 15);
  const int mb = (lane >> 4) * 8;
  #pragma unroll
  for (int j = 0; j < 8; ++j) {
    int row = m0 + mb + j;
    if (row < M) {
      float v = acc[j] + (bias ? bias[col] : 0.f);
      if (mode == 1) {
        const float* xs = (row < split) ? (X1 + (size_t)row * 128)
                                        : (X2 + (size_t)(row - split) * 128);
        v = s * v + om * xs[col];
      }
      C[(size_t)row * 128 + col] = v;
    }
  }
}

// C[M x N] = A[M x 128] @ B[N x 128]^T  (similarity scores)
__global__ __launch_bounds__(256) void bsg_gemm_nt128(
    const float* __restrict__ A, const float* __restrict__ B,
    float* __restrict__ C, int M, int N) {
  __shared__ unsigned short Bs[128 * 128];
  __shared__ unsigned short As[16 * 128];
  const int tid = threadIdx.x;
  const int m0 = blockIdx.x * 16;
  const int n0 = blockIdx.y * 128;
  // B rows are contiguous in k; iterate k innermost for coalesced reads.
  for (int idx = tid; idx < 128 * 128; idx += 256) {
    int col = idx >> 7, kf = idx & 127;
    int kk = kf >> 5, kr = kf & 31;
    int cn = n0 + col;
    Bs[kk * 4096 + col * 32 + kr] =
        bsg_f2bf(cn < N ? B[(size_t)cn * 128 + kf] : 0.f);
  }
  for (int idx = tid; idx < 16 * 128; idx += 256) {
    int r = idx >> 7, c = idx & 127;
    int row = m0 + r;
    As[idx] = bsg_f2bf(row < M ? A[(size_t)row * 128 + c] : 0.f);
  }
  __syncthreads();
  const int lane = tid & 31;
  const int col0 = (tid >> 5) * 16;
  v8f acc = bsg_wmma_tile(As, Bs, lane, col0);
  const int col = n0 + col0 + (lane & 15);
  const int mb = (lane >> 4) * 8;
  #pragma unroll
  for (int j = 0; j < 8; ++j) {
    int row = m0 + mb + j;
    if (row < M && col < N) C[(size_t)row * N + col] = acc[j];
  }
}

__global__ __launch_bounds__(256) void bsg_fill_f32(float* p, float v, size_t n) {
  size_t i = (size_t)blockIdx.x * 256 + threadIdx.x;
  if (i < n) p[i] = v;
}
__global__ __launch_bounds__(256) void bsg_fill_u32(unsigned* p, unsigned v, size_t n) {
  size_t i = (size_t)blockIdx.x * 256 + threadIdx.x;
  if (i < n) p[i] = v;
}

// logit[e,h] = (q[dst,h,:] . (k[src,h,:] @ ar[h])) * pr[h]/sqrt(D); atomicMax per (dst,h)
__global__ __launch_bounds__(256) void bsg_edge_logit(
    const float* __restrict__ K, const float* __restrict__ Q,
    const int* __restrict__ srcI, const int* __restrict__ dstI, int E,
    const float* __restrict__ ar, const float* __restrict__ pr,
    float* __restrict__ logit, unsigned int* __restrict__ maxU) {
  __shared__ float arS[HC * DC * DC];
  for (int idx = threadIdx.x; idx < HC * DC * DC; idx += 256) arS[idx] = ar[idx];
  __syncthreads();
  size_t gid = (size_t)blockIdx.x * 256 + threadIdx.x;
  if (gid >= (size_t)E * HC) return;
  int e = (int)(gid >> 2), h = (int)(gid & 3);
  int s = srcI[e], d = dstI[e];
  const float* kv = K + (size_t)s * DIMC + h * DC;
  const float* qv = Q + (size_t)d * DIMC + h * DC;
  const float* a = arS + h * DC * DC;
  float kr[DC];
  #pragma unroll
  for (int i = 0; i < DC; ++i) kr[i] = kv[i];
  float acc = 0.f;
  for (int f = 0; f < DC; ++f) {
    float t = 0.f;
    #pragma unroll
    for (int dd = 0; dd < DC; ++dd) t += kr[dd] * a[dd * DC + f];
    acc += qv[f] * t;
  }
  float lg = acc * pr[h] * 0.17677669529663687f;  // 1/sqrt(32)
  logit[(size_t)e * HC + h] = lg;
  atomicMax(maxU + (size_t)d * HC + h, bsg_encf(lg));
}

__global__ __launch_bounds__(256) void bsg_edge_exp(
    const int* __restrict__ dstI, int E, float* __restrict__ logit,
    const unsigned int* __restrict__ maxU, float* __restrict__ den) {
  size_t gid = (size_t)blockIdx.x * 256 + threadIdx.x;
  if (gid >= (size_t)E * HC) return;
  int e = (int)(gid >> 2), h = (int)(gid & 3);
  int d = dstI[e];
  float mx = bsg_decf(maxU[(size_t)d * HC + h]);
  float ex = expf(logit[(size_t)e * HC + h] - mx);
  logit[(size_t)e * HC + h] = ex;
  atomicAdd(den + (size_t)d * HC + h, ex);
}

// agg[dst,h,:] += w * (v[src,h,:] @ mr[h])
__global__ __launch_bounds__(256) void bsg_edge_agg(
    const float* __restrict__ V, const int* __restrict__ srcI,
    const int* __restrict__ dstI, int E, const float* __restrict__ mr,
    const float* __restrict__ logit, const float* __restrict__ den,
    float* __restrict__ agg) {
  __shared__ float mrS[HC * DC * DC];
  for (int idx = threadIdx.x; idx < HC * DC * DC; idx += 256) mrS[idx] = mr[idx];
  __syncthreads();
  size_t gid = (size_t)blockIdx.x * 256 + threadIdx.x;
  if (gid >= (size_t)E * HC) return;
  int e = (int)(gid >> 2), h = (int)(gid & 3);
  int s = srcI[e], d = dstI[e];
  float w = logit[(size_t)e * HC + h] /
            fmaxf(den[(size_t)d * HC + h], 1e-16f);
  const float* vv = V + (size_t)s * DIMC + h * DC;
  const float* m = mrS + h * DC * DC;
  float vr[DC];
  #pragma unroll
  for (int i = 0; i < DC; ++i) vr[i] = vv[i];
  float* out = agg + (size_t)d * DIMC + h * DC;
  for (int f = 0; f < DC; ++f) {
    float t = 0.f;
    #pragma unroll
    for (int dd = 0; dd < DC; ++dd) t += vr[dd] * m[dd * DC + f];
    atomicAdd(out + f, w * t);
  }
}

__global__ __launch_bounds__(256) void bsg_combine_relu(
    const float* __restrict__ a, const float* __restrict__ b,
    float* __restrict__ o, size_t n) {
  size_t i = (size_t)blockIdx.x * 256 + threadIdx.x;
  if (i < n) o[i] = fmaxf(0.f, 0.5f * (a[i] + b[i]));
}

__global__ __launch_bounds__(256) void bsg_pool_accum(
    const float* __restrict__ xa, const int* __restrict__ fd,
    const int* __restrict__ fi, int Nd, int Ng, float* __restrict__ sums,
    float* __restrict__ cnt, float* __restrict__ colsum) {
  size_t gid = (size_t)blockIdx.x * 256 + threadIdx.x;
  if (gid >= (size_t)Ng * DIMC) return;
  int n = (int)(gid >> 7), c = (int)(gid & 127);
  int f = (n < Nd) ? fd[n] : fi[n - Nd];
  float v = xa[gid];
  atomicAdd(&sums[(size_t)f * DIMC + c], v);
  atomicAdd(&colsum[c], v);
  if (c == 0) atomicAdd(&cnt[f], 1.f);
}

__global__ __launch_bounds__(128) void bsg_ctx_att(
    const float* __restrict__ colsum, int Ng, float* __restrict__ catt) {
  int c = threadIdx.x;
  catt[c] = tanhf(colsum[c] / (float)Ng);
}
__global__ __launch_bounds__(256) void bsg_ctx_func(
    const float* __restrict__ sums, const float* __restrict__ cnt,
    float* __restrict__ cf) {
  size_t gid = (size_t)blockIdx.x * 256 + threadIdx.x;
  if (gid >= (size_t)FC * DIMC) return;
  cf[gid] = tanhf(sums[gid] / fmaxf(cnt[gid >> 7], 1.f));
}

__global__ __launch_bounds__(256) void bsg_pool_scores(
    const float* __restrict__ xd, const float* __restrict__ xi,
    const int* __restrict__ fd, const int* __restrict__ fi, int Nd, int Ng,
    const float* __restrict__ catt, const float* __restrict__ cf,
    float* __restrict__ patt, float* __restrict__ fp) {
  int n = blockIdx.x * 256 + threadIdx.x;
  if (n >= Ng) return;
  const float* x = (n < Nd) ? (xd + (size_t)n * DIMC)
                            : (xi + (size_t)(n - Nd) * DIMC);
  int f = (n < Nd) ? fd[n] : fi[n - Nd];
  const float* cfr = cf + (size_t)f * DIMC;
  float da = 0.f, df = 0.f;
  for (int c = 0; c < DIMC; ++c) { float xv = x[c]; da += xv * catt[c]; df += xv * cfr[c]; }
  float sa = bsg_sigm(da), sf = bsg_sigm(df);
  for (int c = 0; c < DIMC; ++c) {
    float xv = x[c];
    atomicAdd(patt + c, sa * xv);
    atomicAdd(fp + (size_t)f * DIMC + c, sf * xv);
  }
}

__global__ __launch_bounds__(256) void bsg_minmax(
    const float* __restrict__ sc, size_t n, unsigned* __restrict__ mm) {
  size_t i = (size_t)blockIdx.x * 256 + threadIdx.x;
  if (i >= n) return;
  unsigned u = bsg_encf(sc[i]);
  atomicMin(&mm[0], u);
  atomicMax(&mm[1], u);
}
__global__ __launch_bounds__(256) void bsg_hist(
    const float* __restrict__ sc, size_t n, const unsigned* __restrict__ mm,
    float* __restrict__ hist) {
  size_t i = (size_t)blockIdx.x * 256 + threadIdx.x;
  if (i >= n) return;
  float mn = bsg_decf(mm[0]), mx = bsg_decf(mm[1]);
  float d = fmaxf(mx - mn, 1e-12f);
  int b = (int)floorf((sc[i] - mn) / d * (float)BINSC);
  b = b < 0 ? 0 : (b > BINSC - 1 ? BINSC - 1 : b);
  atomicAdd(&hist[b], 1.f);
}

__global__ __launch_bounds__(128) void bsg_ntn(
    const float* __restrict__ p1, const float* __restrict__ p2,
    const float* __restrict__ Wtn, float* __restrict__ scr) {
  __shared__ float red[128];
  int t = blockIdx.x, j = threadIdx.x;
  float acc = 0.f;
  for (int i = 0; i < DIMC; ++i) acc += p1[i] * Wtn[((size_t)i * DIMC + j) * TC + t];
  red[j] = acc * p2[j];
  __syncthreads();
  for (int s = 64; s > 0; s >>= 1) {
    if (j < s) red[j] += red[j + s];
    __syncthreads();
  }
  if (j == 0) scr[t] = red[0];
}

__global__ __launch_bounds__(32) void bsg_head(
    const float* __restrict__ scr, const float* __restrict__ hist,
    const float* __restrict__ p1, const float* __restrict__ p2,
    const float* __restrict__ Vtn, const float* __restrict__ btn,
    const float* __restrict__ Wfc1, const float* __restrict__ bfc1,
    const float* __restrict__ Wsc, const float* __restrict__ bsc,
    float* __restrict__ out) {
  __shared__ float feats[TC + BINSC];
  __shared__ float hh[16];
  __shared__ float hs;
  int tid = threadIdx.x;
  if (tid < TC) {
    float a = btn[tid];
    for (int i = 0; i < 2 * DIMC; ++i)
      a += Vtn[tid * 2 * DIMC + i] * (i < DIMC ? p1[i] : p2[i - DIMC]);
    feats[tid] = fmaxf(0.f, scr[tid] + a);
  }
  if (tid == 16) {
    float s = 0.f;
    for (int b = 0; b < BINSC; ++b) s += hist[b];
    hs = fmaxf(s, 1.f);
  }
  __syncthreads();
  if (tid < BINSC) feats[TC + tid] = hist[tid] / hs;
  __syncthreads();
  if (tid < 16) {
    float a = bfc1[tid];
    for (int i = 0; i < TC + BINSC; ++i) a += feats[i] * Wfc1[i * 16 + tid];
    hh[tid] = fmaxf(0.f, a);
  }
  __syncthreads();
  if (tid == 0) {
    float a = bsc[0];
    for (int j = 0; j < 16; ++j) a += hh[j] * Wsc[j];
    out[0] = bsg_sigm(a);
  }
}

extern "C" void kernel_launch(void* const* d_in, const int* in_sizes, int n_in,
                              void* d_out, int out_size, void* d_ws, size_t ws_size,
                              hipStream_t stream) {
  (void)out_size; (void)ws_size;
  if (n_in < 36) return;
  const int Nd = in_sizes[0] / DIMC;
  const int Ni = in_sizes[1] / DIMC;
  const int Ng = Nd + Ni;

  const float* Wk = (const float*)d_in[16];
  const float* bk = (const float*)d_in[17];
  const float* Wq = (const float*)d_in[18];
  const float* bq = (const float*)d_in[19];
  const float* Wv = (const float*)d_in[20];
  const float* bv = (const float*)d_in[21];
  const float* Wa = (const float*)d_in[22];
  const float* ba = (const float*)d_in[23];
  const float* skip = (const float*)d_in[24];
  const float* a_rel = (const float*)d_in[25];
  const float* m_rel = (const float*)d_in[26];
  const float* p_rel = (const float*)d_in[27];
  const float* W_att = (const float*)d_in[28];
  const float* W_tn = (const float*)d_in[29];
  const float* V_tn = (const float*)d_in[30];
  const float* b_tn = (const float*)d_in[31];
  const float* W_fc1 = (const float*)d_in[32];
  const float* b_fc1 = (const float*)d_in[33];
  const float* W_sc = (const float*)d_in[34];
  const float* b_sc = (const float*)d_in[35];

  int Ecs[2][4];
  int Esum[2] = {0, 0};
  for (int g = 0; g < 2; ++g)
    for (int r = 0; r < 4; ++r) {
      Ecs[g][r] = in_sizes[g * 8 + 2 + r] / 2;
      Esum[g] += Ecs[g][r];
    }
  size_t Emax = (size_t)(Esum[0] > Esum[1] ? Esum[0] : Esum[1]);

  float* ws = (float*)d_ws;
  size_t off = 0;
  auto alloc = [&](size_t n) { size_t o = off; off += n; return o; };
  const size_t nD = (size_t)Nd * DIMC, nI = (size_t)Ni * DIMC;
  const size_t XA_D = alloc(nD), XA_I = alloc(nI);
  const size_t G_D = alloc(nD), G_I = alloc(nI);
  const size_t C1_D = alloc(nD), C1_I = alloc(nI);
  const size_t C2_D = alloc(nD), C2_I = alloc(nI);
  const size_t K_D = alloc(nD), K_I = alloc(nI);
  const size_t Q_D = alloc(nD), Q_I = alloc(nI);
  const size_t V_D = alloc(nD), V_I = alloc(nI);
  const size_t AG_D = alloc(nD), AG_I = alloc(nI);
  const size_t LG = alloc(Emax * HC);
  const size_t MX_D = alloc((size_t)Nd * HC), MX_I = alloc((size_t)Ni * HC);
  const size_t DN_D = alloc((size_t)Nd * HC), DN_I = alloc((size_t)Ni * HC);
  const size_t SUMS = alloc((size_t)FC * DIMC);
  const size_t CNT = alloc(FC);
  const size_t COLS = alloc(DIMC), CATT = alloc(DIMC);
  const size_t CFB = alloc((size_t)FC * DIMC);
  const size_t PA0 = alloc(DIMC), PA1 = alloc(DIMC);
  const size_t FP0 = alloc((size_t)FC * DIMC), FP1 = alloc((size_t)FC * DIMC);
  const size_t SCR = alloc(TC), HIST = alloc(BINSC), MM = alloc(2);
  const size_t XATT = C1_D;  // alias: conv scratch free during pooling
  const size_t SCb = C2_D;   // alias: conv scratch free after both graphs

  auto fillf = [&](size_t o, float v, size_t n) {
    bsg_fill_f32<<<dim3((unsigned)((n + 255) / 256)), 256, 0, stream>>>(ws + o, v, n);
  };
  auto fillu = [&](size_t o, unsigned v, size_t n) {
    bsg_fill_u32<<<dim3((unsigned)((n + 255) / 256)), 256, 0, stream>>>(
        (unsigned*)(ws + o), v, n);
  };
  auto gemm = [&](const float* A1, int split, const float* A2, const float* W,
                  const float* b, float* C, int M, int mode, const float* sp,
                  const float* X1, const float* X2) {
    bsg_gemm128<<<dim3((unsigned)((M + 15) / 16)), 256, 0, stream>>>(
        A1, split, A2, W, b, C, M, mode, sp, X1, X2);
  };

  const size_t PAg[2] = {PA0, PA1};
  const size_t FPg[2] = {FP0, FP1};

  for (int g = 0; g < 2; ++g) {
    const float* x0d = (const float*)d_in[g * 8 + 0];
    const float* x0i = (const float*)d_in[g * 8 + 1];
    const int* eiP[4] = {(const int*)d_in[g * 8 + 2], (const int*)d_in[g * 8 + 3],
                         (const int*)d_in[g * 8 + 4], (const int*)d_in[g * 8 + 5]};
    const int* fd = (const int*)d_in[g * 8 + 6];
    const int* fi = (const int*)d_in[g * 8 + 7];

    for (int l = 0; l < 2; ++l) {
      const float* xd = (l == 0) ? x0d : ws + XA_D;
      const float* xi = (l == 0) ? x0i : ws + XA_I;
      float* od = (l == 0) ? ws + XA_D : ws + G_D;
      float* oi = (l == 0) ? ws + XA_I : ws + G_I;
      const float* xs[2] = {xd, xi};
      const int Ns[2] = {Nd, Ni};
      float* Kt[2] = {ws + K_D, ws + K_I};
      float* Qt[2] = {ws + Q_D, ws + Q_I};
      float* Vt[2] = {ws + V_D, ws + V_I};
      float* At[2] = {ws + AG_D, ws + AG_I};

      for (int dir = 0; dir < 2; ++dir) {
        const int lin = l * 2 + dir;
        float* Ct[2] = {ws + (dir == 0 ? C1_D : C2_D), ws + (dir == 0 ? C1_I : C2_I)};

        for (int t = 0; t < 2; ++t) {
          size_t wo = (size_t)(lin * 2 + t) * DIMC * DIMC;
          size_t bo = (size_t)(lin * 2 + t) * DIMC;
          gemm(xs[t], Ns[t], nullptr, Wk + wo, bk + bo, Kt[t], Ns[t], 0, nullptr, nullptr, nullptr);
          gemm(xs[t], Ns[t], nullptr, Wq + wo, bq + bo, Qt[t], Ns[t], 0, nullptr, nullptr, nullptr);
          gemm(xs[t], Ns[t], nullptr, Wv + wo, bv + bo, Vt[t], Ns[t], 0, nullptr, nullptr, nullptr);
        }
        fillu(MX_D, 0x007FFFFFu, (size_t)Nd * HC);  // enc(-inf)
        fillu(MX_I, 0x007FFFFFu, (size_t)Ni * HC);
        fillf(DN_D, 0.f, (size_t)Nd * HC);
        fillf(DN_I, 0.f, (size_t)Ni * HC);
        fillf(AG_D, 0.f, nD + nI);  // AG_D and AG_I are contiguous

        struct RelD { const int* s; const int* d; int E; int st; int dt; } rel[4];
        const int* E0 = eiP[0]; const int* E1 = eiP[1];
        const int* E2 = eiP[2]; const int* E3 = eiP[3];
        int e0 = Ecs[g][0], e1 = Ecs[g][1], e2 = Ecs[g][2], e3 = Ecs[g][3];
        if (dir == 0) {
          rel[0] = {E0, E0 + e0, e0, 1, 1};
          rel[1] = {E1, E1 + e1, e1, 0, 1};
          rel[2] = {E2, E2 + e2, e2, 1, 0};
          rel[3] = {E3, E3 + e3, e3, 1, 1};
        } else {
          rel[0] = {E0 + e0, E0, e0, 1, 1};
          rel[1] = {E1 + e1, E1, e1, 1, 0};
          rel[2] = {E2 + e2, E2, e2, 0, 1};
          rel[3] = {E3 + e3, E3, e3, 1, 1};
        }
        size_t loff[4]; size_t cum = 0;
        for (int r = 0; r < 4; ++r) { loff[r] = cum; cum += (size_t)rel[r].E; }

        for (int r = 0; r < 4; ++r) {
          unsigned blk = (unsigned)(((size_t)rel[r].E * HC + 255) / 256);
          bsg_edge_logit<<<dim3(blk), 256, 0, stream>>>(
              Kt[rel[r].st], Qt[rel[r].dt], rel[r].s, rel[r].d, rel[r].E,
              a_rel + (size_t)(lin * 4 + r) * HC * DC * DC,
              p_rel + (size_t)(lin * 4 + r) * HC,
              ws + LG + loff[r] * HC,
              (unsigned*)(ws + (rel[r].dt ? MX_I : MX_D)));
        }
        for (int r = 0; r < 4; ++r) {
          unsigned blk = (unsigned)(((size_t)rel[r].E * HC + 255) / 256);
          bsg_edge_exp<<<dim3(blk), 256, 0, stream>>>(
              rel[r].d, rel[r].E, ws + LG + loff[r] * HC,
              (const unsigned*)(ws + (rel[r].dt ? MX_I : MX_D)),
              ws + (rel[r].dt ? DN_I : DN_D));
        }
        for (int r = 0; r < 4; ++r) {
          unsigned blk = (unsigned)(((size_t)rel[r].E * HC + 255) / 256);
          bsg_edge_agg<<<dim3(blk), 256, 0, stream>>>(
              Vt[rel[r].st], rel[r].s, rel[r].d, rel[r].E,
              m_rel + (size_t)(lin * 4 + r) * HC * DC * DC,
              ws + LG + loff[r] * HC, ws + (rel[r].dt ? DN_I : DN_D),
              ws + (rel[r].dt ? AG_I : AG_D));
        }
        for (int t = 0; t < 2; ++t) {
          size_t wo = (size_t)(lin * 2 + t) * DIMC * DIMC;
          size_t bo = (size_t)(lin * 2 + t) * DIMC;
          gemm(At[t], Ns[t], nullptr, Wa + wo, ba + bo, Ct[t], Ns[t], 1,
               skip + (size_t)(lin * 2 + t), xs[t], nullptr);
        }
      }
      bsg_combine_relu<<<dim3((unsigned)((nD + 255) / 256)), 256, 0, stream>>>(
          ws + C1_D, ws + C2_D, od, nD);
      bsg_combine_relu<<<dim3((unsigned)((nI + 255) / 256)), 256, 0, stream>>>(
          ws + C1_I, ws + C2_I, oi, nI);
    }

    // ---- pooling for this graph ----
    fillf(SUMS, 0.f, (size_t)FC * DIMC);
    fillf(CNT, 0.f, FC);
    fillf(COLS, 0.f, DIMC);
    fillf(PAg[g], 0.f, DIMC);
    fillf(FPg[g], 0.f, (size_t)FC * DIMC);
    gemm(ws + G_D, Nd, ws + G_I, W_att, nullptr, ws + XATT, Ng, 0, nullptr, nullptr, nullptr);
    bsg_pool_accum<<<dim3((unsigned)(((size_t)Ng * DIMC + 255) / 256)), 256, 0, stream>>>(
        ws + XATT, fd, fi, Nd, Ng, ws + SUMS, ws + CNT, ws + COLS);
    bsg_ctx_att<<<1, 128, 0, stream>>>(ws + COLS, Ng, ws + CATT);
    bsg_ctx_func<<<dim3((unsigned)(((size_t)FC * DIMC + 255) / 256)), 256, 0, stream>>>(
        ws + SUMS, ws + CNT, ws + CFB);
    bsg_pool_scores<<<dim3((unsigned)((Ng + 255) / 256)), 256, 0, stream>>>(
        ws + G_D, ws + G_I, fd, fi, Nd, Ng, ws + CATT, ws + CFB,
        ws + PAg[g], ws + FPg[g]);
  }

  // ---- similarity histogram + NTN + head ----
  bsg_gemm_nt128<<<dim3((unsigned)((FC + 15) / 16), (unsigned)((FC + 127) / 128)),
                   256, 0, stream>>>(ws + FP0, ws + FP1, ws + SCb, FC, FC);
  fillu(MM, 0xFFFFFFFFu, 1);
  fillu(MM + 1, 0u, 1);
  size_t nsc = (size_t)FC * FC;
  bsg_minmax<<<dim3((unsigned)((nsc + 255) / 256)), 256, 0, stream>>>(
      ws + SCb, nsc, (unsigned*)(ws + MM));
  fillf(HIST, 0.f, BINSC);
  bsg_hist<<<dim3((unsigned)((nsc + 255) / 256)), 256, 0, stream>>>(
      ws + SCb, nsc, (const unsigned*)(ws + MM), ws + HIST);
  bsg_ntn<<<dim3(TC), 128, 0, stream>>>(ws + PA0, ws + PA1, W_tn, ws + SCR);
  bsg_head<<<1, 32, 0, stream>>>(ws + SCR, ws + HIST, ws + PA0, ws + PA1,
                                 V_tn, b_tn, W_fc1, b_fc1, W_sc, b_sc,
                                 (float*)d_out);
}
